// MultiHeadAttentionV2_81561428951836
// MI455X (gfx1250) — matrix operand order repaired and checked by reading, not compiled, not run on hardware
//
#include <hip/hip_runtime.h>
#include <cstdint>

typedef _Float16 h16;
typedef __attribute__((ext_vector_type(8)))  _Float16 v8h;
typedef __attribute__((ext_vector_type(16))) _Float16 v16h;
typedef __attribute__((ext_vector_type(8)))  float    v8f;

// ---- LDS byte-offset of a generic pointer to __shared__ (low 32 bits) ----
static __device__ inline unsigned lds_off(const void* p) {
    return (unsigned)reinterpret_cast<uintptr_t>(p);
}

// ---- async global->LDS copy of 32 bytes (2 x b128); ASYNCcnt-tracked ----
static __device__ inline void async_cp32B(unsigned lds, const h16* g) {
    uint64_t ga = (uint64_t)reinterpret_cast<uintptr_t>(g);
    asm volatile("global_load_async_to_lds_b128 %0, %1, off"
                 :: "v"(lds), "v"(ga) : "memory");
    // INST_OFFSET applies to both LDS and global sides (ISA 08 section 4.4)
    asm volatile("global_load_async_to_lds_b128 %0, %1, off offset:16"
                 :: "v"(lds), "v"(ga) : "memory");
}

static __device__ inline void wait_async0() {
#if __has_builtin(__builtin_amdgcn_s_wait_asynccnt)
    __builtin_amdgcn_s_wait_asynccnt(0);
#else
    asm volatile("s_wait_asynccnt 0x0" ::: "memory");
#endif
}

// =====================================================================
// Attention GEMM (f16 x f16): C[m][n] = scale * sum_k A[m][k]*Bsrc[n][k]
// Both operands N-major/K-contiguous -> staged with async global->LDS
// (ASYNCcnt) into double-buffered LDS with compile-time buffer indices;
// WMMA on one buffer overlaps the async DMA filling the other.
// =====================================================================
template <int BUF>
static __device__ inline void attn_tile(const h16 (&As)[2][128][40],
                                        const h16 (&Bs)[2][128][40],
                                        v8f (&acc)[2][4],
                                        int wm, int wn, int lhalf, int l15)
{
    v16h af[2], bf[4];
#pragma unroll
    for (int i = 0; i < 2; ++i) {
        const int r = wm * 32 + i * 16 + l15;
        v8h lo = *reinterpret_cast<const v8h*>(&As[BUF][r][8 * lhalf]);
        v8h hi = *reinterpret_cast<const v8h*>(&As[BUF][r][8 * lhalf + 16]);
        af[i] = __builtin_shufflevector(lo, hi, 0,1,2,3,4,5,6,7,8,9,10,11,12,13,14,15);
    }
#pragma unroll
    for (int j = 0; j < 4; ++j) {
        const int c = wn * 64 + j * 16 + l15;
        v8h lo = *reinterpret_cast<const v8h*>(&Bs[BUF][c][8 * lhalf]);
        v8h hi = *reinterpret_cast<const v8h*>(&Bs[BUF][c][8 * lhalf + 16]);
        bf[j] = __builtin_shufflevector(lo, hi, 0,1,2,3,4,5,6,7,8,9,10,11,12,13,14,15);
    }
#pragma unroll
    for (int i = 0; i < 2; ++i)
#pragma unroll
        for (int j = 0; j < 4; ++j)
            acc[i][j] = __builtin_amdgcn_wmma_f32_16x16x32_f16(
                false, af[i], false, bf[j], (short)0, acc[i][j], false, false);
}

template <typename TOUT, bool SCALE>
__global__ __launch_bounds__(256)
void attn_gemm(const h16* __restrict__ A, const h16* __restrict__ Bsrc,
               TOUT* __restrict__ C, int K, int lda, int ldb, int ldc,
               long sA, long sB, long sC, float scale)
{
    __shared__ alignas(16) h16 As[2][128][40];
    __shared__ alignas(16) h16 Bs[2][128][40];

    const long z = blockIdx.z;
    A    += z * sA;
    Bsrc += z * sB;
    C    += z * sC;

    const int m0   = blockIdx.y * 128;
    const int n0   = blockIdx.x * 128;
    const int tid  = threadIdx.x;
    const int lane = tid & 31;
    const int wid  = tid >> 5;
    const int wm   = wid & 3;
    const int wn   = wid >> 2;
    const int lhalf = lane >> 4;
    const int l15   = lane & 15;
    const int sm = tid >> 1;
    const int sk = (tid & 1) * 16;

    const h16* aship = A    + (long)(m0 + sm) * lda + sk;   // +k0 per step
    const h16* bship = Bsrc + (long)(n0 + sm) * ldb + sk;
    const unsigned aoff[2] = { lds_off(&As[0][sm][sk]), lds_off(&As[1][sm][sk]) };
    const unsigned boff[2] = { lds_off(&Bs[0][sm][sk]), lds_off(&Bs[1][sm][sk]) };

    v8f acc[2][4] = {};

    // prologue: DMA first K tile into buffer 0
    async_cp32B(aoff[0], aship);
    async_cp32B(boff[0], bship);
    wait_async0();
    __syncthreads();

    // K is a multiple of 64 for all uses (1024, 4096)
    for (int k0 = 0; k0 < K; k0 += 64) {
        // phase 0: DMA k0+32 into buf1 while computing buf0
        async_cp32B(aoff[1], aship + k0 + 32);
        async_cp32B(boff[1], bship + k0 + 32);
        attn_tile<0>(As, Bs, acc, wm, wn, lhalf, l15);
        wait_async0();
        __syncthreads();

        // phase 1: DMA k0+64 into buf0 while computing buf1
        if (k0 + 64 < K) {
            async_cp32B(aoff[0], aship + k0 + 64);
            async_cp32B(boff[0], bship + k0 + 64);
        }
        attn_tile<1>(As, Bs, acc, wm, wn, lhalf, l15);
        wait_async0();
        __syncthreads();
    }

#pragma unroll
    for (int i = 0; i < 2; ++i)
#pragma unroll
        for (int j = 0; j < 4; ++j) {
            const int n  = n0 + wn * 64 + j * 16 + l15;
            const int mb = m0 + wm * 32 + i * 16 + 8 * lhalf;
#pragma unroll
            for (int v = 0; v < 8; ++v) {
                float val = acc[i][j][v];
                if constexpr (SCALE) val *= scale;
                C[(long)(mb + v) * ldc + n] = (TOUT)val;
            }
        }
}

// force device emission here, ahead of the other kernels
template __global__ void attn_gemm<h16, true>(const h16*, const h16*, h16*,
    int, int, int, int, long, long, long, float);
template __global__ void attn_gemm<float, false>(const h16*, const h16*, float*,
    int, int, int, int, long, long, long, float);

// ---- 16-element f32 load + convert-to-f16 (projection staging) ----
static __device__ inline void cvt16(h16* dst, const float* __restrict__ src) {
    const float4* p = reinterpret_cast<const float4*>(src);
#pragma unroll
    for (int i = 0; i < 4; ++i) {
        float4 f = p[i];
        dst[i*4+0] = (h16)f.x; dst[i*4+1] = (h16)f.y;
        dst[i*4+2] = (h16)f.z; dst[i*4+3] = (h16)f.w;
    }
}

// =====================================================================
// Projection GEMM: C[m][n] = sum_k A[m][k]*W[n][k] + bias[n]
// f32 inputs (converted to f16 while staging), f16 output.
// =====================================================================
__global__ __launch_bounds__(256)
void proj_gemm(const float* __restrict__ A, const float* __restrict__ W,
               const float* __restrict__ bias, h16* __restrict__ C,
               int K, int lda, int ldb, int ldc)
{
    __shared__ alignas(16) h16 As[128][40];
    __shared__ alignas(16) h16 Bs[128][40];

    const int m0   = blockIdx.y * 128;
    const int n0   = blockIdx.x * 128;
    const int tid  = threadIdx.x;
    const int lane = tid & 31;
    const int wid  = tid >> 5;
    const int wm   = wid & 3;
    const int wn   = wid >> 2;
    const int lhalf = lane >> 4;
    const int l15   = lane & 15;
    const int sm = tid >> 1;
    const int sk = (tid & 1) * 16;

    v8f acc[2][4] = {};

    for (int k0 = 0; k0 < K; k0 += 32) {
        __syncthreads();
        {
            h16 tmp[16];
            cvt16(tmp, A + (long)(m0 + sm) * lda + k0 + sk);
            *reinterpret_cast<v8h*>(&As[sm][sk])     = *reinterpret_cast<v8h*>(tmp);
            *reinterpret_cast<v8h*>(&As[sm][sk + 8]) = *reinterpret_cast<v8h*>(tmp + 8);
            cvt16(tmp, W + (long)(n0 + sm) * ldb + k0 + sk);
            *reinterpret_cast<v8h*>(&Bs[sm][sk])     = *reinterpret_cast<v8h*>(tmp);
            *reinterpret_cast<v8h*>(&Bs[sm][sk + 8]) = *reinterpret_cast<v8h*>(tmp + 8);
        }
        __syncthreads();

        if (k0 + 32 < K) {  // global_prefetch_b8 next K tile
            __builtin_prefetch(A + (long)(m0 + sm) * lda + (k0 + 32) + sk, 0, 1);
            __builtin_prefetch(W + (long)(n0 + sm) * ldb + (k0 + 32) + sk, 0, 1);
        }

        v16h af[2], bf[4];
#pragma unroll
        for (int i = 0; i < 2; ++i) {
            const int r = wm * 32 + i * 16 + l15;
            v8h lo = *reinterpret_cast<const v8h*>(&As[r][8 * lhalf]);
            v8h hi = *reinterpret_cast<const v8h*>(&As[r][8 * lhalf + 16]);
            af[i] = __builtin_shufflevector(lo, hi, 0,1,2,3,4,5,6,7,8,9,10,11,12,13,14,15);
        }
#pragma unroll
        for (int j = 0; j < 4; ++j) {
            const int c = wn * 64 + j * 16 + l15;
            v8h lo = *reinterpret_cast<const v8h*>(&Bs[c][8 * lhalf]);
            v8h hi = *reinterpret_cast<const v8h*>(&Bs[c][8 * lhalf + 16]);
            bf[j] = __builtin_shufflevector(lo, hi, 0,1,2,3,4,5,6,7,8,9,10,11,12,13,14,15);
        }
#pragma unroll
        for (int i = 0; i < 2; ++i)
#pragma unroll
            for (int j = 0; j < 4; ++j)
                acc[i][j] = __builtin_amdgcn_wmma_f32_16x16x32_f16(
                    false, af[i], false, bf[j], (short)0, acc[i][j], false, false);
    }

#pragma unroll
    for (int i = 0; i < 2; ++i)
#pragma unroll
        for (int j = 0; j < 4; ++j) {
            const int n  = n0 + wn * 64 + j * 16 + l15;
            const int mb = m0 + wm * 32 + i * 16 + 8 * lhalf;
            const float bv = bias[n];
#pragma unroll
            for (int v = 0; v < 8; ++v)
                C[(long)(mb + v) * ldc + n] = (h16)(acc[i][j][v] + bv);
        }
}

// =====================================================================
// 64x64 tiled f16 transpose: out[c][r] = in[r][c]  (per batch z)
// =====================================================================
__global__ __launch_bounds__(256)
void transpose_h16(const h16* __restrict__ in, h16* __restrict__ outp,
                   int ldi, int ldo, long sIn, long sOut)
{
    __shared__ alignas(16) h16 t[64][72];
    in   += blockIdx.z * sIn;
    outp += blockIdx.z * sOut;
    const int r0 = blockIdx.y * 64;
    const int c0 = blockIdx.x * 64;
    const int tid = threadIdx.x;

    {   // coalesced load 64x64 tile
        const int lr = tid >> 2;
        const int lc = (tid & 3) * 16;
        const h16* p = in + (long)(r0 + lr) * ldi + c0 + lc;
        *reinterpret_cast<v8h*>(&t[lr][lc])     = *reinterpret_cast<const v8h*>(p);
        *reinterpret_cast<v8h*>(&t[lr][lc + 8]) = *reinterpret_cast<const v8h*>(p + 8);
    }
    __syncthreads();
    {   // coalesced store of transposed tile
        const int oc = tid >> 2;          // original column
        const int os = (tid & 3) * 16;    // original row segment
        v8h a, b;
#pragma unroll
        for (int e = 0; e < 8; ++e) { a[e] = t[os + e][oc]; b[e] = t[os + 8 + e][oc]; }
        h16* p = outp + (long)(c0 + oc) * ldo + r0 + os;
        *reinterpret_cast<v8h*>(p)     = a;
        *reinterpret_cast<v8h*>(p + 8) = b;
    }
}

// =====================================================================
// In-place row softmax on f16 matrix; one 256-thread block per row.
// =====================================================================
__global__ __launch_bounds__(256)
void softmax_rows(h16* __restrict__ S, int ncols)
{
    h16* rp = S + (long)blockIdx.x * ncols;
    const int tid  = threadIdx.x;
    const int lane = tid & 31;
    const int wid  = tid >> 5;

    float x[16];
    {
        v8h a = *reinterpret_cast<const v8h*>(rp + tid * 16);
        v8h b = *reinterpret_cast<const v8h*>(rp + tid * 16 + 8);
#pragma unroll
        for (int e = 0; e < 8; ++e) { x[e] = (float)a[e]; x[8 + e] = (float)b[e]; }
    }

    __shared__ float red[8];
    __shared__ float bcast[2];

    float mx = -1e30f;
#pragma unroll
    for (int e = 0; e < 16; ++e) mx = fmaxf(mx, x[e]);
    for (int off = 16; off > 0; off >>= 1) mx = fmaxf(mx, __shfl_xor(mx, off, 32));
    if (lane == 0) red[wid] = mx;
    __syncthreads();
    if (tid == 0) {
        float m = red[0];
        for (int i = 1; i < 8; ++i) m = fmaxf(m, red[i]);
        bcast[0] = m;
    }
    __syncthreads();
    mx = bcast[0];

    float s = 0.f;
#pragma unroll
    for (int e = 0; e < 16; ++e) { x[e] = __expf(x[e] - mx); s += x[e]; }
    for (int off = 16; off > 0; off >>= 1) s += __shfl_xor(s, off, 32);
    __syncthreads();
    if (lane == 0) red[wid] = s;
    __syncthreads();
    if (tid == 0) {
        float t = 0.f;
        for (int i = 0; i < 8; ++i) t += red[i];
        bcast[1] = 1.f / t;
    }
    __syncthreads();
    const float inv = bcast[1];

    v8h oa, ob;
#pragma unroll
    for (int e = 0; e < 8; ++e) { oa[e] = (h16)(x[e] * inv); ob[e] = (h16)(x[8 + e] * inv); }
    *reinterpret_cast<v8h*>(rp + tid * 16)     = oa;
    *reinterpret_cast<v8h*>(rp + tid * 16 + 8) = ob;
}

// =====================================================================
extern "C" void kernel_launch(void* const* d_in, const int* in_sizes, int n_in,
                              void* d_out, int out_size, void* d_ws, size_t ws_size,
                              hipStream_t stream)
{
    constexpr int Bb = 2, S = 4096, D = 1024;
    constexpr long SD = (long)S * D;
    constexpr long SS = (long)S * S;

    const float* xq = (const float*)d_in[0];
    const float* xk = (const float*)d_in[1];
    const float* xv = (const float*)d_in[2];
    const float* Wq = (const float*)d_in[3];
    const float* bq = (const float*)d_in[4];
    const float* Wk = (const float*)d_in[5];
    const float* bk = (const float*)d_in[6];
    const float* Wv = (const float*)d_in[7];
    const float* bv = (const float*)d_in[8];
    float* out = (float*)d_out;

    h16* Qh = (h16*)d_ws;            // [B*S, D] f16
    h16* Kh = Qh + Bb * SD;
    h16* Vh = Kh + Bb * SD;
    h16* Vt = Vh + Bb * SD;          // [B, D, S] f16 (V transposed)
    h16* Sc = Vt + Bb * SD;          // [B, S, S] f16 (scores -> probs)

    const dim3 blk(256);
    const float scale = 0.03125f;    // 1/sqrt(1024)

    // 1) QKV projections (f32 in, f16 out)
    const dim3 gp(D / 128, (Bb * S) / 128, 1);
    proj_gemm<<<gp, blk, 0, stream>>>(xq, Wq, bq, Qh, D, D, D, D);
    proj_gemm<<<gp, blk, 0, stream>>>(xk, Wk, bk, Kh, D, D, D, D);
    proj_gemm<<<gp, blk, 0, stream>>>(xv, Wv, bv, Vh, D, D, D, D);

    // 2) transpose V: Vt[b][d][s] = Vh[b][s][d]
    const dim3 gt(D / 64, S / 64, Bb);
    transpose_h16<<<gt, blk, 0, stream>>>(Vh, Vt, D, S, SD, SD);

    // 3) scores = (Q @ K^T) * scale  (async-staged WMMA, f16 out)
    const dim3 gs(S / 128, S / 128, Bb);
    attn_gemm<h16, true><<<gs, blk, 0, stream>>>(
        Qh, Kh, Sc, D, D, D, S, SD, SD, SS, scale);

    // 4) softmax rows, in place
    softmax_rows<<<dim3(Bb * S), blk, 0, stream>>>(Sc, S);

    // 5) context = P @ V = P @ Vt^T  (async-staged WMMA, f32 out)
    const dim3 gc(D / 128, S / 128, Bb);
    attn_gemm<float, false><<<gc, blk, 0, stream>>>(
        Sc, Vt, out, S, S, S, D, SS, SD, SD, 1.f);
}
